// Model_15908558865528
// MI455X (gfx1250) — matrix operand order repaired and checked by reading, not compiled, not run on hardware
//
#include <hip/hip_runtime.h>
#include <hip/hip_bf16.h>
#include <stdint.h>

// ---------------------------------------------------------------------------
// N=1024, D_IN=512, D=128, drop p=0.4
//   q,k,v,scale = x_i @ W^T + b ; smul = 1/softmax(scale, axis=0)
//   qk = q@k^T ; P[n,d,m] = dropout(softmax_m(qk[n,m]*smul[n,d]))
//   out[n,d,kk] = sum_m P[n,d,m] * v[m,kk]
// GEMMs: v_wmma_f32_16x16x32_f16 (wave32). v streamed to LDS with
// GLOBAL_LOAD_ASYNC_TO_LDS_B128 (ASYNCcnt) when the builtin is available,
// double-buffered so the copy overlaps the WMMA work on the previous chunk.
// ---------------------------------------------------------------------------

typedef __attribute__((ext_vector_type(16))) _Float16 v16h;
typedef __attribute__((ext_vector_type(8)))  _Float16 v8h;
typedef __attribute__((ext_vector_type(8)))  float    v8f;
typedef int v4i32 __attribute__((vector_size(16)));
typedef __attribute__((address_space(1))) v4i32* as1_v4i32_p;  // global int4*
typedef __attribute__((address_space(3))) v4i32* as3_v4i32_p;  // LDS int4*

#define NN    1024
#define DIN   512
#define DD    128
#define KEEP_SCALE 1.6666666f   // 1/(1-0.4)
#define KEEP_THRESH 2576980378u // 0.6 * 2^32

#if __has_builtin(__builtin_amdgcn_global_load_async_to_lds_b128) && \
    __has_builtin(__builtin_amdgcn_s_wait_asynccnt)
#define USE_ASYNC_LDS 1
#else
#define USE_ASYNC_LDS 0
#endif

// PCG-ish hash for deterministic dropout keyed on flat (n,d,m) index
__device__ __forceinline__ uint32_t drop_hash(uint32_t idx) {
  uint32_t h = idx * 747796405u + 2891336453u;
  h = ((h >> ((h >> 28) + 4u)) ^ h) * 277803737u;
  return (h >> 22) ^ h;
}

__device__ __forceinline__ v16h pack16(float4 a, float4 b, float4 c, float4 d) {
  v16h r;
  r[0]  = (_Float16)a.x; r[1]  = (_Float16)a.y; r[2]  = (_Float16)a.z; r[3]  = (_Float16)a.w;
  r[4]  = (_Float16)b.x; r[5]  = (_Float16)b.y; r[6]  = (_Float16)b.z; r[7]  = (_Float16)b.w;
  r[8]  = (_Float16)c.x; r[9]  = (_Float16)c.y; r[10] = (_Float16)c.z; r[11] = (_Float16)c.w;
  r[12] = (_Float16)d.x; r[13] = (_Float16)d.y; r[14] = (_Float16)d.z; r[15] = (_Float16)d.w;
  return r;
}

// ---------------------------------------------------------------------------
// Stage 1: y = x_i @ W^T + b for i=0..3. One wave per 16x16 tile.
// grid (N/16, D/16, 4), block 32.
//   which 0 -> qh (f16 [N][D]) ; 1 -> kh (f16 [N][D])
//   which 2 -> vhT (f16 TRANSPOSED [D][N]) ; 3 -> scalef (f32 [N][D])
// ---------------------------------------------------------------------------
__global__ __launch_bounds__(32)
void lin_kernel(const float* __restrict__ x1, const float* __restrict__ x2,
                const float* __restrict__ x3, const float* __restrict__ x4,
                const float* __restrict__ W,  const float* __restrict__ bias,
                _Float16* __restrict__ qh, _Float16* __restrict__ kh,
                _Float16* __restrict__ vhT, float* __restrict__ scalef)
{
  const int lane = threadIdx.x;
  const int ln   = lane & 15;
  const int row0 = blockIdx.x * 16;
  const int col0 = blockIdx.y * 16;
  const int which = blockIdx.z;
  const float* x = (which == 0) ? x1 : (which == 1) ? x2 : (which == 2) ? x3 : x4;

  const int aoff = (lane >> 4) << 3;  // A half-lane k base (0 / 8)
  const int boff = (lane >> 4) << 3;  // B half-lane k base (0 / 8)

  v8f acc = {};
  for (int k0 = 0; k0 < DIN; k0 += 32) {
    const float* px = x + (size_t)(row0 + ln) * DIN + k0 + aoff;
    const float* pw = W + (size_t)(col0 + ln) * DIN + k0 + boff;
    v16h a  = pack16(*(const float4*)px,        *(const float4*)(px + 4),
                     *(const float4*)(px + 16), *(const float4*)(px + 20));
    v16h bf = pack16(*(const float4*)pw,        *(const float4*)(pw + 4),
                     *(const float4*)(pw + 16), *(const float4*)(pw + 20));
    acc = __builtin_amdgcn_wmma_f32_16x16x32_f16(false, a, false, bf,
                                                 (short)0, acc, false, false);
  }
  const float bv = bias[col0 + ln];

  if (which == 3) {
#pragma unroll
    for (int r = 0; r < 8; ++r) {
      const int m = ((lane >> 4) << 3) + r;
      scalef[(size_t)(row0 + m) * DD + col0 + ln] = acc[r] + bv;
    }
  } else if (which == 2) {
#pragma unroll
    for (int r = 0; r < 8; ++r) {
      const int m = ((lane >> 4) << 3) + r;
      vhT[(size_t)(col0 + ln) * NN + row0 + m] = (_Float16)(acc[r] + bv);
    }
  } else {
    _Float16* yh = (which == 0) ? qh : kh;
#pragma unroll
    for (int r = 0; r < 8; ++r) {
      const int m = ((lane >> 4) << 3) + r;
      yh[(size_t)(row0 + m) * DD + col0 + ln] = (_Float16)(acc[r] + bv);
    }
  }
}

// ---------------------------------------------------------------------------
// Stage 2: smul[n,d] = 1/softmax(scale, axis=0) = sumExp * exp(colmax - x)
// ---------------------------------------------------------------------------
__global__ __launch_bounds__(256)
void colsoftmax_kernel(const float* __restrict__ scalef, float* __restrict__ smul)
{
  __shared__ float red[256];
  const int d = blockIdx.x, t = threadIdx.x;
  float mx = -3e38f;
  for (int n = t; n < NN; n += 256) mx = fmaxf(mx, scalef[n * DD + d]);
  red[t] = mx; __syncthreads();
  for (int s = 128; s > 0; s >>= 1) {
    if (t < s) red[t] = fmaxf(red[t], red[t + s]);
    __syncthreads();
  }
  mx = red[0]; __syncthreads();
  float sum = 0.0f;
  for (int n = t; n < NN; n += 256) sum += __expf(scalef[n * DD + d] - mx);
  red[t] = sum; __syncthreads();
  for (int s = 128; s > 0; s >>= 1) {
    if (t < s) red[t] += red[t + s];
    __syncthreads();
  }
  sum = red[0]; __syncthreads();
  for (int n = t; n < NN; n += 256)
    smul[n * DD + d] = sum * __expf(mx - scalef[n * DD + d]);
}

// ---------------------------------------------------------------------------
// Stage 3: qk = q @ k^T (f32). grid (64,64), block 32. Fragments via b128.
// ---------------------------------------------------------------------------
__global__ __launch_bounds__(32)
void qk_kernel(const _Float16* __restrict__ qh, const _Float16* __restrict__ kh,
               float* __restrict__ qk)
{
  const int lane = threadIdx.x, ln = lane & 15;
  const int r0 = blockIdx.x * 16;
  const int c0 = blockIdx.y * 16;
  const int aoff = (lane >> 4) << 3;
  const int boff = (lane >> 4) << 3;
  v8f acc = {};
  for (int k0 = 0; k0 < DD; k0 += 32) {
    const _Float16* pa = qh + (size_t)(r0 + ln) * DD + k0 + aoff;
    const _Float16* pb = kh + (size_t)(c0 + ln) * DD + k0 + boff;
    v8h a0 = *(const v8h*)pa, a1 = *(const v8h*)(pa + 16);
    v8h b0 = *(const v8h*)pb, b1 = *(const v8h*)(pb + 16);
    v16h a  = __builtin_shufflevector(a0, a1, 0,1,2,3,4,5,6,7,8,9,10,11,12,13,14,15);
    v16h bf = __builtin_shufflevector(b0, b1, 0,1,2,3,4,5,6,7,8,9,10,11,12,13,14,15);
    acc = __builtin_amdgcn_wmma_f32_16x16x32_f16(false, a, false, bf,
                                                 (short)0, acc, false, false);
  }
#pragma unroll
  for (int r = 0; r < 8; ++r) {
    const int m = ((lane >> 4) << 3) + r;
    qk[(size_t)(r0 + m) * NN + c0 + ln] = acc[r];
  }
}

// ---------------------------------------------------------------------------
// Chunk stage: vhT[:, m0:m0+32] -> buf ([col][32] f16). 512 x 16B, 2/thread.
// Async path: GLOBAL_LOAD_ASYNC_TO_LDS_B128 (no VGPR transit, ASYNCcnt).
// ---------------------------------------------------------------------------
__device__ __forceinline__ void stage_chunk(const _Float16* __restrict__ vhT,
                                            _Float16* buf, int m0, int t)
{
#pragma unroll
  for (int j = 0; j < 2; ++j) {
    const int c   = t * 2 + j;       // 0..511
    const int col = c >> 2;
    const int off = (c & 3) * 8;
    const _Float16* src = vhT + (size_t)col * NN + m0 + off;
    _Float16* dst = buf + col * 32 + off;
#if USE_ASYNC_LDS
    __builtin_amdgcn_global_load_async_to_lds_b128(
        (as1_v4i32_p)(void*)src,
        (as3_v4i32_p)(void*)dst, 0, 0);
#else
    *(v8h*)dst = *(const v8h*)src;
#endif
  }
}

__device__ __forceinline__ void chunk_wait()
{
#if USE_ASYNC_LDS
  __builtin_amdgcn_s_wait_asynccnt(0);
#endif
  __syncthreads();
}

// ---------------------------------------------------------------------------
// Stage 4 (fused): per n -- row max, softmax denominators, dropout, P@V.
// 256 threads = 8 waves; wave w owns d-tile w with 8 kk-tile accumulators.
// v double-buffered through LDS; B fragment = one 32B LDS read per lane.
// ---------------------------------------------------------------------------
__global__ __launch_bounds__(256)
void pv_kernel(const float* __restrict__ qk, const float* __restrict__ smul,
               const _Float16* __restrict__ vhT, float* __restrict__ out)
{
  __shared__ float    qrow[NN];           // 4 KB
  __shared__ float    red[256];           // 1 KB
  __shared__ float    sm[DD];             // 512 B
  __shared__ float    invden[DD];         // 512 B
  __shared__ _Float16 vbufT[2][DD * 32];  // 2 x 8 KB, [col][m] per chunk

  const int n = blockIdx.x;
  const int t = threadIdx.x;

  for (int m = t; m < NN; m += 256) qrow[m] = qk[(size_t)n * NN + m];
  if (t < DD) sm[t] = smul[n * DD + t];
  // Kick off chunk 0 of v while the reductions run
  stage_chunk(vhT, vbufT[0], 0, t);
  __syncthreads();

  // Row max M_n over qrow
  float mx = -3e38f;
  for (int m = t; m < NN; m += 256) mx = fmaxf(mx, qrow[m]);
  red[t] = mx; __syncthreads();
  for (int s = 128; s > 0; s >>= 1) {
    if (t < s) red[t] = fmaxf(red[t], red[t + s]);
    __syncthreads();
  }
  const float Mn = red[0];

  // Denominators: thread (d = t&127, half = t>>7) sums 512 exps
  {
    const int d = t & 127, half = t >> 7;
    const float s = sm[d];
    float sum = 0.0f;
    const int mend = half * 512 + 512;
    for (int m = half * 512; m < mend; ++m) sum += __expf(s * (qrow[m] - Mn));
    __syncthreads();                   // red[0] consumed by all before reuse
    red[t] = sum; __syncthreads();
    if (t < DD) invden[t] = 1.0f / (red[t] + red[t + 128]);
  }
  chunk_wait();                        // invden visible + chunk 0 landed

  const int wave = t >> 5;
  const int lane = t & 31;
  const int ln   = lane & 15;
  const int d0   = wave * 16;
  const int drow = d0 + ln;            // A-fragment row = this lane's d
  const float srow = sm[drow];
  const float invd = invden[drow];
  const int aoff = (lane >> 4) << 4;   // 0 / 16
  const int boff = (lane >> 4) << 4;   // 0 / 16

  v8f acc[8] = {};

  for (int m0 = 0; m0 < NN; m0 += 32) {
    const int cur = (m0 >> 5) & 1;
    if (m0 + 32 < NN)                  // stream next chunk into other buffer
      stage_chunk(vhT, vbufT[cur ^ 1], m0 + 32, t);

    // A tile: dropout-masked softmax probs P[d, m0+.]; qrow via b128 reads
    float qv[16];
    const float* qp = qrow + m0 + aoff;
    *(float4*)&qv[0]  = *(const float4*)(qp);
    *(float4*)&qv[4]  = *(const float4*)(qp + 4);
    *(float4*)&qv[8]  = *(const float4*)(qp + 8);
    *(float4*)&qv[12] = *(const float4*)(qp + 12);

    v16h a;
#pragma unroll
    for (int e = 0; e < 16; ++e) {
      const int m = m0 + aoff + e;
      float p = __expf(srow * (qv[e] - Mn)) * invd;
      const uint32_t idx = ((uint32_t)n << 17) | ((uint32_t)drow << 10) | (uint32_t)m;
      p = (drop_hash(idx) < KEEP_THRESH) ? p * KEEP_SCALE : 0.0f;
      a[e] = (_Float16)p;
    }

#pragma unroll
    for (int kt = 0; kt < 8; ++kt) {
      const int col = kt * 16 + ln;
      const v16h bf = *(const v16h*)&vbufT[cur][col * 32 + boff]; // 32B aligned
      acc[kt] = __builtin_amdgcn_wmma_f32_16x16x32_f16(false, a, false, bf,
                                                       (short)0, acc[kt],
                                                       false, false);
    }
    chunk_wait();                      // next buffer landed; cur free for reuse
  }

#pragma unroll
  for (int kt = 0; kt < 8; ++kt) {
#pragma unroll
    for (int r = 0; r < 8; ++r) {
      const int d = d0 + ((lane >> 4) << 3) + r;
      out[((size_t)n * DD + d) * DD + kt * 16 + ln] = acc[kt][r];
    }
  }
}

// ---------------------------------------------------------------------------
// Host-side launcher. Inputs: x1,x2,x3,x4,W,b. Output: [1024,128,128] f32.
// ---------------------------------------------------------------------------
extern "C" void kernel_launch(void* const* d_in, const int* in_sizes, int n_in,
                              void* d_out, int out_size, void* d_ws, size_t ws_size,
                              hipStream_t stream)
{
  (void)in_sizes; (void)n_in; (void)out_size; (void)ws_size;
  const float* x1 = (const float*)d_in[0];
  const float* x2 = (const float*)d_in[1];
  const float* x3 = (const float*)d_in[2];
  const float* x4 = (const float*)d_in[3];
  const float* W  = (const float*)d_in[4];
  const float* b  = (const float*)d_in[5];
  float* out = (float*)d_out;

  char* ws = (char*)d_ws;
  _Float16* qh     = (_Float16*)(ws + 0);        // 256 KB, [N][D]
  _Float16* kh     = (_Float16*)(ws + 262144);   // 256 KB, [N][D]
  _Float16* vhT    = (_Float16*)(ws + 524288);   // 256 KB, [D][N] (transposed)
  float*    scalef = (float*)   (ws + 786432);   // 512 KB
  float*    smul   = (float*)   (ws + 1310720);  // 512 KB
  float*    qk     = (float*)   (ws + 1835008);  // 4 MB

  lin_kernel<<<dim3(NN / 16, DD / 16, 4), 32, 0, stream>>>(
      x1, x2, x3, x4, W, b, qh, kh, vhT, scalef);
  colsoftmax_kernel<<<DD, 256, 0, stream>>>(scalef, smul);
  qk_kernel<<<dim3(NN / 16, NN / 16), 32, 0, stream>>>(qh, kh, qk);
  pv_kernel<<<NN, 256, 0, stream>>>(qk, smul, vhT, out);
}
